// RoutedTransformerLayer_64450279244307
// MI455X (gfx1250) — matrix operand order repaired and checked by reading, not compiled, not run on hardware
//
#include <hip/hip_runtime.h>
#include <hip/hip_bf16.h>

typedef __attribute__((ext_vector_type(16))) _Float16 v16h;
typedef __attribute__((ext_vector_type(8)))  _Float16 v8h;
typedef __attribute__((ext_vector_type(8)))  float    v8f;
typedef unsigned int u32x4 __attribute__((ext_vector_type(4)));
typedef int          i32x8 __attribute__((ext_vector_type(8)));
typedef int          i32x4 __attribute__((ext_vector_type(4)));

#define BSZ   4
#define SEQ   1024
#define HDIM  1024
#define MTOK  (BSZ*SEQ)      // 4096 tokens
#define NHEADS 16
#define DH     64
#define QKV_LD 3072
#define IDIM  4096
#define RHDIM 256
#define NPATH 16
#define TOPK  4

#define EPI_BIAS    1
#define EPI_RELU    2
#define EPI_GELU    4
#define EPI_RWSCALE 8

// ---------------------------------------------------------------- helpers

__device__ __forceinline__ v8f wmma16(v16h a, v16h b, v8f c) {
  return __builtin_amdgcn_wmma_f32_16x16x32_f16(
      /*neg_a=*/false, a, /*neg_b=*/false, b,
      /*c_mod=*/(short)0, c, /*reuse_a=*/false, /*reuse_b=*/false);
}

// A-fragment (16x32 f16, MxK): lane row = lane&15, kb = 8*(lane>>4),
// elements j<8 -> K=kb+j ; j>=8 -> K=kb+16+(j-8)
__device__ __forceinline__ v16h load_a_frag(const _Float16* rowptr, int kbase) {
  v8h c0 = *(const v8h*)(rowptr + kbase);
  v8h c1 = *(const v8h*)(rowptr + kbase + 16);
  v16h a;
#pragma unroll
  for (int j = 0; j < 8; ++j) { a[j] = c0[j]; a[j + 8] = c1[j]; }
  return a;
}

__device__ __forceinline__ float gelu_tanh(float x) {
  float x3 = x * x * x;
  return 0.5f * x * (1.0f + tanhf(0.7978845608f * (x + 0.044715f * x3)));
}

// ---- Tensor Data Mover: stage a 64(row) x 32(k) f16 tile into LDS.
// D# per CDNA5 ISA: group0 = {count/flags, lds_addr,
// global_addr lo, global_addr hi | type<<30}; group1 packs data_size=2B,
// tensor dims (large, never OOB), tile_dim0=32, tile_dim1=64, dim0 stride.
__device__ __forceinline__ void tdm_load_b_tile(const _Float16* gsrc,
                                                unsigned lds_off,
                                                unsigned ldb_elems) {
  unsigned long long ga = (unsigned long long)(uintptr_t)gsrc;
  u32x4 g0;
  g0[0] = 1u;                                            // count=1, user mode
  g0[1] = lds_off;                                       // LDS byte address
  g0[2] = (unsigned)(ga & 0xffffffffu);                  // global addr [31:0]
  g0[3] = (unsigned)((ga >> 32) & 0x01ffffffu) | (2u << 30); // [56:32] | type=2
  const unsigned td0 = 1u << 20, td1 = 1u << 20;         // tensor dims (no OOB)
  i32x8 g1;
  g1[0] = (int)(1u << 16);                               // wg_mask=0, data_size=2B
  g1[1] = (int)((td0 & 0xffffu) << 16);                  // barrier_addr=0 | td0.lo
  g1[2] = (int)((td0 >> 16) | ((td1 & 0xffffu) << 16));  // td0.hi | td1.lo
  g1[3] = (int)((td1 >> 16) | (32u << 16));              // td1.hi | tile_dim0=32
  g1[4] = 64;                                            // tile_dim1=64, tile_dim2=0
  g1[5] = (int)ldb_elems;                                // dim0 stride [31:0]
  g1[6] = 0;                                             // stride hi | dim1 stride lo
  g1[7] = 0;
  i32x4 z4 = {0, 0, 0, 0};
#if defined(__clang_major__) && (__clang_major__ >= 23)
  i32x8 z8 = {0, 0, 0, 0, 0, 0, 0, 0};
  __builtin_amdgcn_tensor_load_to_lds(g0, g1, z4, z4, z8, 0);
#else
  __builtin_amdgcn_tensor_load_to_lds(g0, g1, z4, z4, 0);
#endif
}

// ---------------------------------------------------------------- convert f32 -> f16

__global__ void __launch_bounds__(256) cvt_kernel(const float* __restrict__ s,
                                                  _Float16* __restrict__ d, int n) {
  int i = blockIdx.x * blockDim.x + threadIdx.x;
  int str = gridDim.x * blockDim.x;
  for (; i < n; i += str) d[i] = (_Float16)s[i];
}

// ---------------------------------------------------------------- layernorm (row = token)

__global__ void __launch_bounds__(256) ln_kernel(const float* __restrict__ x,
                                                 const float* __restrict__ w,
                                                 const float* __restrict__ b,
                                                 _Float16* __restrict__ out) {
  const int row = blockIdx.x, tid = threadIdx.x;
  const float* xr = x + (size_t)row * HDIM;
  float lx[4], s = 0.f;
#pragma unroll
  for (int i = 0; i < 4; ++i) { lx[i] = xr[tid + i * 256]; s += lx[i]; }
  __shared__ float red[256];
  red[tid] = s; __syncthreads();
  for (int st = 128; st > 0; st >>= 1) { if (tid < st) red[tid] += red[tid + st]; __syncthreads(); }
  const float mean = red[0] * (1.f / HDIM);
  __syncthreads();
  float s2 = 0.f;
#pragma unroll
  for (int i = 0; i < 4; ++i) { float d = lx[i] - mean; s2 += d * d; }
  red[tid] = s2; __syncthreads();
  for (int st = 128; st > 0; st >>= 1) { if (tid < st) red[tid] += red[tid + st]; __syncthreads(); }
  const float rstd = rsqrtf(red[0] * (1.f / HDIM) + 1e-5f);
#pragma unroll
  for (int i = 0; i < 4; ++i) {
    int c = tid + i * 256;
    out[(size_t)row * HDIM + c] = (_Float16)((lx[i] - mean) * rstd * w[c] + b[c]);
  }
}

// ---------------------------------------------------------------- f16 WMMA GEMM, TDM-staged B
// C[M,N] = A[M,K] @ W[N,K]^T (+epilogue). Block = 8 waves; wave -> 32x64 tile
// (two A fragments reuse every B fragment). The 64x32 B tile shared by all 8
// waves is DMA'd into LDS by the Tensor Data Mover, double-buffered:
// wave0 issues tensor_load_to_lds for K-step i+1, everyone computes step i
// from LDS, wave0 s_wait_tensorcnt(0), block barrier, swap buffers.
// Requires M % 256 == 0, N % 64 == 0, K % 32 == 0 (all call sites comply).

__global__ void __launch_bounds__(256) gemm_f16(
    const _Float16* __restrict__ A, int lda, long long pA,
    const _Float16* __restrict__ W, int ldb, long long pW,
    const float* __restrict__ bias, long long pBias,
    _Float16* __restrict__ outh, float* __restrict__ outf, int ldc, long long pOut,
    const float* __restrict__ resid, int ldr, long long pResid,
    const float* __restrict__ rowW,
    int M, int N, int K, int epi) {
  const int lane = threadIdx.x & 31;
  const int wv   = threadIdx.x >> 5;
  const int half = lane >> 4;
  const int l15  = lane & 15;
  const int z    = blockIdx.z;
  const int mbase = (blockIdx.y * 8 + wv) * 32;
  const int nbase = blockIdx.x * 64;

  __shared__ _Float16 Bsh[2][64][32] __attribute__((aligned(128)));

  A += (size_t)z * pA;  W += (size_t)z * pW;
  if (bias)  bias  += (size_t)z * pBias;
  if (resid) resid += (size_t)z * pResid;
  if (outh)  outh  += (size_t)z * pOut;
  if (outf)  outf  += (size_t)z * pOut;

  const _Float16* arow0 = A + (size_t)(mbase + l15) * lda;
  const _Float16* arow1 = A + (size_t)(mbase + 16 + l15) * lda;
  const _Float16* wtile = W + (size_t)nbase * ldb;
  const int kb = half * 8;
  const int kh = half * 16;

  // prologue: stage first B tile
  if (wv == 0) {
    tdm_load_b_tile(wtile, (unsigned)(uintptr_t)&Bsh[0][0][0], (unsigned)ldb);
    __builtin_amdgcn_s_wait_tensorcnt(0);
  }
  __syncthreads();

  v8f acc[2][4] = {};
  int cur = 0;
  for (int k0 = 0; k0 < K; k0 += 32) {
    if (wv == 0 && (k0 + 32) < K)  // prefetch next K-step into the other buffer
      tdm_load_b_tile(wtile + k0 + 32,
                      (unsigned)(uintptr_t)&Bsh[cur ^ 1][0][0], (unsigned)ldb);

    v16h a0 = load_a_frag(arow0, k0 + kb);
    v16h a1 = load_a_frag(arow1, k0 + kb);
#pragma unroll
    for (int nt = 0; nt < 4; ++nt) {
      v16h b = load_a_frag(&Bsh[cur][nt * 16 + l15][0], kb);
      acc[0][nt] = wmma16(a0, b, acc[0][nt]);
      acc[1][nt] = wmma16(a1, b, acc[1][nt]);
    }

    if (wv == 0 && (k0 + 32) < K) __builtin_amdgcn_s_wait_tensorcnt(0);
    __syncthreads();   // reads of Bsh[cur] done everywhere; Bsh[cur^1] filled
    cur ^= 1;
  }

  float rws[2][8];
  if (epi & EPI_RWSCALE) {
#pragma unroll
    for (int ms = 0; ms < 2; ++ms)
#pragma unroll
      for (int r = 0; r < 8; ++r)
        rws[ms][r] = rowW[(size_t)(mbase + ms * 16 + r + half * 8) * NPATH + z];
  }

#pragma unroll
  for (int nt = 0; nt < 4; ++nt) {
    const int col = nbase + nt * 16 + l15;
    const float bb = (epi & EPI_BIAS) ? bias[col] : 0.f;
#pragma unroll
    for (int ms = 0; ms < 2; ++ms) {
#pragma unroll
      for (int r = 0; r < 8; ++r) {
        float v = acc[ms][nt][r] + bb;
        if (epi & EPI_RELU) v = fmaxf(v, 0.f);
        if (epi & EPI_GELU) v = gelu_tanh(v);
        if (epi & EPI_RWSCALE) v *= rws[ms][r];
        const int mrow = mbase + ms * 16 + r + half * 8;
        if (resid) v += resid[(size_t)mrow * ldr + col];
        if (outf) outf[(size_t)mrow * ldc + col] = v;
        if (outh) outh[(size_t)mrow * ldc + col] = (_Float16)v;
      }
    }
  }
}

// ---------------------------------------------------------------- flash attention
// one wave per (batch*head, 16-query tile); 32 keys / iteration; online softmax.

__global__ void __launch_bounds__(128) attn_kernel(const _Float16* __restrict__ qkv,
                                                   _Float16* __restrict__ ctx) {
  const int lane = threadIdx.x & 31;
  const int w    = threadIdx.x >> 5;
  const int half = lane >> 4;
  const int l15  = lane & 15;
  const int qt   = blockIdx.x;            // query tile (same for all 4 waves -> barrier-safe)
  const int bh   = blockIdx.y * 4 + w;
  const int b    = bh >> 4;
  const int hh   = bh & 15;

  __shared__ _Float16 Psh[4][16][32] __attribute__((aligned(64)));

  const _Float16* qrow = qkv + (size_t)(b * SEQ + qt * 16 + l15) * QKV_LD + hh * DH;
  const int kb = half * 8;
  const v16h aq0 = *(const v16h*)(qrow + half * 16);        // d 0..31
  const v16h aq1 = *(const v16h*)(qrow + 32 + half * 16);   // d 32..63

  v8f oacc[4] = {{}, {}, {}, {}};
  float mrow_[8], lrow_[8];
#pragma unroll
  for (int r = 0; r < 8; ++r) { mrow_[r] = -1e30f; lrow_[r] = 0.f; }

  const int nk = (qt + 2) >> 1;                  // ceil((qt+1)*16 / 32)
  for (int it = 0; it < nk; ++it) {
    const int k0 = it * 32;
    // ---- scores: S = Q K^T * 1/sqrt(Dh)
    v8f s0 = {}, s1 = {};
    {
      const _Float16* kr0 = qkv + (size_t)(b * SEQ + k0 + l15) * QKV_LD + HDIM + hh * DH;
      const _Float16* kr1 = qkv + (size_t)(b * SEQ + k0 + 16 + l15) * QKV_LD + HDIM + hh * DH;
      const int dh0 = half * 16;
      v16h bk;
      bk = *(const v16h*)(kr0 + dh0);      s0 = wmma16(aq0, bk, s0);
      bk = *(const v16h*)(kr0 + 32 + dh0); s0 = wmma16(aq1, bk, s0);
      bk = *(const v16h*)(kr1 + dh0);      s1 = wmma16(aq0, bk, s1);
      bk = *(const v16h*)(kr1 + 32 + dh0); s1 = wmma16(aq1, bk, s1);
    }
    // ---- scale + causal mask + online softmax
    float p0[8], p1[8];
#pragma unroll
    for (int r = 0; r < 8; ++r) {
      const int q = qt * 16 + r + half * 8;
      float v0 = s0[r] * 0.125f; if (k0 + l15 > q)      v0 = -1e30f;
      float v1 = s1[r] * 0.125f; if (k0 + 16 + l15 > q) v1 = -1e30f;
      float rm = fmaxf(v0, v1);
      rm = fmaxf(rm, __shfl_xor(rm, 1, 32));
      rm = fmaxf(rm, __shfl_xor(rm, 2, 32));
      rm = fmaxf(rm, __shfl_xor(rm, 4, 32));
      rm = fmaxf(rm, __shfl_xor(rm, 8, 32));
      const float mnew = fmaxf(mrow_[r], rm);
      const float fr = __expf(mrow_[r] - mnew);
      mrow_[r] = mnew;
      p0[r] = __expf(v0 - mnew);
      p1[r] = __expf(v1 - mnew);
      float rs = p0[r] + p1[r];
      rs += __shfl_xor(rs, 1, 32);
      rs += __shfl_xor(rs, 2, 32);
      rs += __shfl_xor(rs, 4, 32);
      rs += __shfl_xor(rs, 8, 32);
      lrow_[r] = lrow_[r] * fr + rs;
      oacc[0][r] *= fr; oacc[1][r] *= fr; oacc[2][r] *= fr; oacc[3][r] *= fr;
    }
    // ---- C-layout -> A-fragment relayout of P through LDS
    __syncthreads();
#pragma unroll
    for (int r = 0; r < 8; ++r) {
      const int mr = r + half * 8;
      Psh[w][mr][l15]      = (_Float16)p0[r];
      Psh[w][mr][16 + l15] = (_Float16)p1[r];
    }
    __syncthreads();
    const v16h aP = *(const v16h*)(&Psh[w][l15][half * 16]);
    // ---- O += P V  (4 d-blocks of 16)
    const _Float16* vbase =
        qkv + (size_t)(b * SEQ + k0 + half * 16) * QKV_LD + 2 * HDIM + hh * DH + l15;
#pragma unroll
    for (int dd = 0; dd < 4; ++dd) {
      v16h bv;
#pragma unroll
      for (int j = 0; j < 16; ++j) bv[j] = vbase[(size_t)j * QKV_LD + dd * 16];
      oacc[dd] = wmma16(aP, bv, oacc[dd]);
    }
  }

  // ---- normalize + store ctx (f16)
#pragma unroll
  for (int r = 0; r < 8; ++r) {
    const float inv = 1.0f / lrow_[r];
    const size_t row = (size_t)(b * SEQ + qt * 16 + r + half * 8);
    _Float16* crow = ctx + row * HDIM + hh * DH + l15;
    crow[0]  = (_Float16)(oacc[0][r] * inv);
    crow[16] = (_Float16)(oacc[1][r] * inv);
    crow[32] = (_Float16)(oacc[2][r] * inv);
    crow[48] = (_Float16)(oacc[3][r] * inv);
  }
}

// ---------------------------------------------------------------- router: logits -> softmax -> top4 -> renorm

__global__ void __launch_bounds__(64) router_kernel(const _Float16* __restrict__ r1,
                                                    const float* __restrict__ Wr2,
                                                    const float* __restrict__ br2,
                                                    float* __restrict__ wgt, int M) {
  const int t = blockIdx.x * blockDim.x + threadIdx.x;
  if (t >= M) return;
  const _Float16* xr = r1 + (size_t)t * RHDIM;
  float lg[NPATH];
#pragma unroll 1
  for (int p = 0; p < NPATH; ++p) {
    float s = br2[p];
    const float* wr = Wr2 + (size_t)p * RHDIM;
    for (int k = 0; k < RHDIM; ++k) s += (float)xr[k] * wr[k];
    lg[p] = s;
  }
  float mx = lg[0];
  for (int p = 1; p < NPATH; ++p) mx = fmaxf(mx, lg[p]);
  float pr[NPATH], sm = 0.f;
  for (int p = 0; p < NPATH; ++p) { pr[p] = __expf(lg[p] - mx); sm += pr[p]; }
  const float ism = 1.f / sm;
  for (int p = 0; p < NPATH; ++p) pr[p] *= ism;
  float keep[NPATH]; bool used[NPATH];
  for (int p = 0; p < NPATH; ++p) { keep[p] = 0.f; used[p] = false; }
  float ssum = 0.f;
  for (int it = 0; it < TOPK; ++it) {   // strict > keeps lowest index on ties (matches top_k)
    int bi = -1; float bv = -1.f;
    for (int p = 0; p < NPATH; ++p)
      if (!used[p] && pr[p] > bv) { bv = pr[p]; bi = p; }
    used[bi] = true; keep[bi] = bv; ssum += bv;
  }
  const float inv = 1.f / (ssum + 1e-8f);
  for (int p = 0; p < NPATH; ++p) wgt[(size_t)t * NPATH + p] = keep[p] * inv;
}

// ---------------------------------------------------------------- launch

extern "C" void kernel_launch(void* const* d_in, const int* in_sizes, int n_in,
                              void* d_out, int out_size, void* d_ws, size_t ws_size,
                              hipStream_t stream) {
  const float* hidden = (const float*)d_in[0];
  const float* ln1w = (const float*)d_in[1];
  const float* ln1b = (const float*)d_in[2];
  const float* Wqkv = (const float*)d_in[3];
  const float* bqkv = (const float*)d_in[4];
  const float* Wo   = (const float*)d_in[5];
  const float* bo   = (const float*)d_in[6];
  const float* ln2w = (const float*)d_in[7];
  const float* ln2b = (const float*)d_in[8];
  const float* Wr1  = (const float*)d_in[9];
  const float* br1  = (const float*)d_in[10];
  const float* Wr2  = (const float*)d_in[11];
  const float* br2  = (const float*)d_in[12];
  const float* W1   = (const float*)d_in[13];
  const float* b1   = (const float*)d_in[14];
  const float* W2   = (const float*)d_in[15];
  const float* b2   = (const float*)d_in[16];
  float* out = (float*)d_out;

  char* ws = (char*)d_ws;
  size_t off = 0;
  auto alloc = [&](size_t bytes) -> void* {
    void* p = ws + off;
    off = (off + bytes + 255) & ~(size_t)255;
    return p;
  };

  _Float16* Wqkv16 = (_Float16*)alloc((size_t)3 * HDIM * HDIM * 2);
  _Float16* Wo16   = (_Float16*)alloc((size_t)HDIM * HDIM * 2);
  _Float16* Wr116  = (_Float16*)alloc((size_t)RHDIM * HDIM * 2);
  _Float16* W116   = (_Float16*)alloc((size_t)IDIM * HDIM * 2);
  _Float16* W216   = (_Float16*)alloc((size_t)HDIM * IDIM * 2);
  _Float16* x16    = (_Float16*)alloc((size_t)MTOK * HDIM * 2);
  _Float16* qkv16  = (_Float16*)alloc((size_t)MTOK * 3 * HDIM * 2);
  _Float16* ctx16  = (_Float16*)alloc((size_t)MTOK * HDIM * 2);
  float*    hbuf   = (float*)alloc((size_t)MTOK * HDIM * 4);
  _Float16* x2_16  = (_Float16*)alloc((size_t)MTOK * HDIM * 2);
  _Float16* r1buf  = (_Float16*)alloc((size_t)MTOK * RHDIM * 2);
  float*    wgt    = (float*)alloc((size_t)MTOK * NPATH * 4);
  _Float16* inter  = (_Float16*)alloc((size_t)NPATH * MTOK * 256 * 2);
  (void)ws_size; (void)in_sizes; (void)n_in; (void)out_size;

  // weight conversion f32 -> f16
  cvt_kernel<<<4096, 256, 0, stream>>>(Wqkv, Wqkv16, 3 * HDIM * HDIM);
  cvt_kernel<<<2048, 256, 0, stream>>>(Wo, Wo16, HDIM * HDIM);
  cvt_kernel<<<512, 256, 0, stream>>>(Wr1, Wr116, RHDIM * HDIM);
  cvt_kernel<<<4096, 256, 0, stream>>>(W1, W116, IDIM * HDIM);
  cvt_kernel<<<4096, 256, 0, stream>>>(W2, W216, HDIM * IDIM);

  // LN1
  ln_kernel<<<MTOK, 256, 0, stream>>>(hidden, ln1w, ln1b, x16);

  // QKV projection: [4096,3072] = x @ Wqkv^T + bqkv
  gemm_f16<<<dim3(3072 / 64, MTOK / 256, 1), 256, 0, stream>>>(
      x16, HDIM, 0, Wqkv16, HDIM, 0, bqkv, 0,
      qkv16, nullptr, QKV_LD, 0, nullptr, 0, 0, nullptr,
      MTOK, 3 * HDIM, HDIM, EPI_BIAS);

  // causal flash attention
  attn_kernel<<<dim3(SEQ / 16, (BSZ * NHEADS) / 4), 128, 0, stream>>>(qkv16, ctx16);

  // output projection + residual: h = hidden + ctx @ Wo^T + bo (f32)
  gemm_f16<<<dim3(HDIM / 64, MTOK / 256, 1), 256, 0, stream>>>(
      ctx16, HDIM, 0, Wo16, HDIM, 0, bo, 0,
      nullptr, hbuf, HDIM, 0, hidden, HDIM, 0, nullptr,
      MTOK, HDIM, HDIM, EPI_BIAS);

  // LN2
  ln_kernel<<<MTOK, 256, 0, stream>>>(hbuf, ln2w, ln2b, x2_16);

  // router MLP stage 1: relu(x2 @ Wr1^T + br1) -> [4096,256] f16
  gemm_f16<<<dim3(RHDIM / 64, MTOK / 256, 1), 256, 0, stream>>>(
      x2_16, HDIM, 0, Wr116, HDIM, 0, br1, 0,
      r1buf, nullptr, RHDIM, 0, nullptr, 0, 0, nullptr,
      MTOK, RHDIM, HDIM, EPI_BIAS | EPI_RELU);

  // router stage 2 + softmax + top-4 + renorm -> wgt[4096,16]
  router_kernel<<<MTOK / 64, 64, 0, stream>>>(r1buf, Wr2, br2, wgt, MTOK);

  // pathway MLP up: inter[p] = gelu(x2[:, p*64:] @ W1b[p]^T + b1b[p]); [16][4096][256]
  gemm_f16<<<dim3(256 / 64, MTOK / 256, NPATH), 256, 0, stream>>>(
      x2_16, HDIM, 64, W116, HDIM, (long long)256 * HDIM + 64, b1, 256,
      inter, nullptr, 256, (long long)MTOK * 256, nullptr, 0, 0, nullptr,
      MTOK, 256, 64, EPI_BIAS | EPI_GELU);

  // pathway MLP down + routing weight + residual:
  // out[:, p*64:] = h[:, p*64:] + (inter[p] @ W2b[p]^T + b2b[p]) * wgt[:, p]
  gemm_f16<<<dim3(1, MTOK / 256, NPATH), 256, 0, stream>>>(
      inter, 256, (long long)MTOK * 256, W216, IDIM, (long long)64 * IDIM + 256, b2, 64,
      nullptr, out, HDIM, 64, hbuf, HDIM, 64, wgt,
      MTOK, 64, 256, EPI_BIAS | EPI_RWSCALE);
}
